// Bigram_5884105195592
// MI455X (gfx1250) — matrix operand order repaired and checked by reading, not compiled, not run on hardware
//
#include <hip/hip_runtime.h>

typedef __bf16 bf16;
typedef bf16  v16bf __attribute__((ext_vector_type(16)));
typedef float v8f   __attribute__((ext_vector_type(8)));

constexpr int   VOCAB   = 65;
constexpr int   NE      = 32;   // n_embd
constexpr int   HD      = 8;    // head dim
constexpr int   WAVES   = 4;
constexpr int   THREADS = WAVES * 32;
constexpr int   NTILES  = 13;   // 6 QKV + 2 MLP + 5 LM
constexpr float SQRT_C  = 5.656854249492381f;   // faithful bug: scores *= sqrt(C)

// Load a 32-byte WMMA fragment (16 bf16) from 16B-aligned LDS: 2x ds_load_b128
__device__ __forceinline__ v16bf load_frag16(const bf16* p) {
  union { v16bf v; float4 q[2]; } u;
  u.q[0] = *(const float4*)(p);
  u.q[1] = *(const float4*)(p + 8);
  return u.v;
}

__global__ __launch_bounds__(THREADS) void fused_fwd(
    const int* __restrict__ idx, const int* __restrict__ tgt,
    const float* __restrict__ tok_emb, const float* __restrict__ pos_emb,
    const float* __restrict__ Wq, const float* __restrict__ Wk, const float* __restrict__ Wv,
    const float* __restrict__ Wm, const float* __restrict__ bm,
    const float* __restrict__ Wl, const float* __restrict__ bl,
    float* __restrict__ logits, float* __restrict__ partials, int n_tok)
{
  // Pre-swizzled B fragments: sBF[tile][lane][elem]; elem e at lane L holds
  // B(K = e + 16*(L>=16), N = L&15) of that 32x16 tile. Tiles 0-5: QKV (N=96),
  // tiles 6-7: MLP (N=32), tiles 8-12: LM head (N=80, vocab zero-padded).
  __shared__ alignas(16) bf16  sBF[NTILES][32][16];   // 13.0 KB
  __shared__ alignas(16) float sF[WAVES][16][96];     // 24 KB: QKV, then logits
  __shared__ alignas(16) bf16  sX[WAVES][16][NE];     // 4 KB : activations
  __shared__ float sbm[NE];
  __shared__ float sbl[80];
  __shared__ float sWL[WAVES];

  const int tid  = threadIdx.x;
  const int wave = tid >> 5;
  const int lane = tid & 31;

  // ---- stage weights once per block, directly into fragment layout ---------
  for (int p = tid; p < NTILES * 32; p += THREADS) {
    const int t  = p >> 5;
    const int L  = p & 31;
    const int kb = (L < 16) ? 0 : 16;
    const int nl = L & 15;
    union { bf16 h[16]; float4 q[2]; } u;
    if (t < 6) {                       // fused QKV, logical column 0..95
      const int col = t * 16 + nl;
      const float* W;
      int cc;
      if (col < 32)      { W = Wq; cc = col; }
      else if (col < 64) { W = Wk; cc = col - 32; }
      else               { W = Wv; cc = col - 64; }
      const int h = cc >> 3, d = cc & 7;
#pragma unroll
      for (int e = 0; e < 16; ++e) u.h[e] = (bf16)W[(h * NE + (kb + e)) * HD + d];
    } else if (t < 8) {                // MLP
      const int n = (t - 6) * 16 + nl;
#pragma unroll
      for (int e = 0; e < 16; ++e) u.h[e] = (bf16)Wm[(kb + e) * NE + n];
    } else {                           // LM head (pad cols >= 65 with zero)
      const int n = (t - 8) * 16 + nl;
#pragma unroll
      for (int e = 0; e < 16; ++e)
        u.h[e] = (bf16)((n < VOCAB) ? Wl[(kb + e) * VOCAB + n] : 0.0f);
    }
    *(float4*)&sBF[t][L][0] = u.q[0];
    *(float4*)&sBF[t][L][8] = u.q[1];
  }
  for (int i = tid; i < NE; i += THREADS) sbm[i] = bm[i];
  for (int i = tid; i < 80; i += THREADS) sbl[i] = (i < VOCAB) ? bl[i] : 0.0f;
  __syncthreads();

  // 16 token rows (= 2 sequences of T=8) per wave
  const long tok_base = ((long)blockIdx.x * WAVES + wave) * 16;
  const bool full_tile = (tok_base + 16 <= (long)n_tok);

  // ---- embedding gather: x = tok_emb[idx] + pos_emb (vectorized) -----------
  {
    const int row = lane & 15;
    const int c0  = (lane < 16) ? 0 : 16;
    long g  = tok_base + row;
    long gc = (g < n_tok) ? g : (long)(n_tok - 1);
    const int id = idx[gc];
    const int t  = row & 7;
    const float4* te = (const float4*)(tok_emb + id * NE + c0);
    const float4* pe = (const float4*)(pos_emb + t * NE + c0);
    union { bf16 h[16]; float4 q[2]; } u;
#pragma unroll
    for (int q4 = 0; q4 < 4; ++q4) {
      float4 a = te[q4], b = pe[q4];
      u.h[q4 * 4 + 0] = (bf16)(a.x + b.x);
      u.h[q4 * 4 + 1] = (bf16)(a.y + b.y);
      u.h[q4 * 4 + 2] = (bf16)(a.z + b.z);
      u.h[q4 * 4 + 3] = (bf16)(a.w + b.w);
    }
    *(float4*)&sX[wave][row][c0]     = u.q[0];
    *(float4*)&sX[wave][row][c0 + 8] = u.q[1];
  }
  __syncthreads();

  // A-fragment loader: lane L<16 -> M=L, K 0-15; L>=16 -> M=L-16, K 16-31
  const int am = lane & 15;
  const int ak = (lane < 16) ? 0 : 16;
  const int rb = (lane < 16) ? 0 : 8;      // C/D row base
  const int cn = lane & 15;                // C/D column-within-tile

#define LOAD_A()                                                     \
  ({ union { v16bf v; float4 q[2]; } _a;                             \
     const bf16* _p = &sX[wave][am][0];                              \
     _a.q[0] = *(const float4*)(_p + ak);                            \
     _a.q[1] = *(const float4*)(_p + ak + 8);                        \
     _a.v; })

  // ---- fused QKV: [16x32] @ [32x96] -> 6 WMMA tiles ------------------------
  {
    v16bf a = LOAD_A();
#pragma unroll
    for (int nt = 0; nt < 6; ++nt) {
      v16bf b = load_frag16(&sBF[nt][lane][0]);
      union { v8f v; float f[8]; } acc;
#pragma unroll
      for (int j = 0; j < 8; ++j) acc.f[j] = 0.0f;
      acc.v = __builtin_amdgcn_wmma_f32_16x16x32_bf16(false, a, false, b,
                                                      (short)0, acc.v, false, false);
      const int col = nt * 16 + cn;
#pragma unroll
      for (int j = 0; j < 8; ++j) sF[wave][rb + j][col] = acc.f[j];
    }
  }
  __syncthreads();

  // ---- attention: per (seq, token, head) causal softmax (8x8, VALU) --------
#pragma unroll
  for (int half = 0; half < 2; ++half) {
    const int pp  = lane + half * 32;       // 64 (seq,token,head) pairs / wave
    const int s   = pp >> 5;
    const int r   = pp & 31;
    const int t   = r >> 2;
    const int h   = r & 3;
    const int row = s * 8 + t;
    union { float f[8]; float4 q[2]; } qv, kv, vv;
    qv.q[0] = *(const float4*)&sF[wave][row][h * 8];
    qv.q[1] = *(const float4*)&sF[wave][row][h * 8 + 4];
    float sc[8];
    float mx = -3.0e38f;
    for (int u = 0; u <= t; ++u) {
      kv.q[0] = *(const float4*)&sF[wave][s * 8 + u][32 + h * 8];
      kv.q[1] = *(const float4*)&sF[wave][s * 8 + u][32 + h * 8 + 4];
      float dot = 0.0f;
#pragma unroll
      for (int d = 0; d < 8; ++d) dot += qv.f[d] * kv.f[d];
      dot *= SQRT_C;
      sc[u] = dot;
      mx = fmaxf(mx, dot);
    }
    float den = 0.0f;
    for (int u = 0; u <= t; ++u) { sc[u] = __expf(sc[u] - mx); den += sc[u]; }
    const float inv = 1.0f / den;
    float o[8];
#pragma unroll
    for (int d = 0; d < 8; ++d) o[d] = 0.0f;
    for (int u = 0; u <= t; ++u) {
      const float w = sc[u] * inv;
      vv.q[0] = *(const float4*)&sF[wave][s * 8 + u][64 + h * 8];
      vv.q[1] = *(const float4*)&sF[wave][s * 8 + u][64 + h * 8 + 4];
#pragma unroll
      for (int d = 0; d < 8; ++d) o[d] += w * vv.f[d];
    }
    union { bf16 h8[8]; float4 q; } ou;
#pragma unroll
    for (int d = 0; d < 8; ++d) ou.h8[d] = (bf16)o[d];
    *(float4*)&sX[wave][row][h * 8] = ou.q;   // head-concat, one b128 store
  }
  __syncthreads();

  // ---- MLP: relu([16x32] @ [32x32] + b) -> 2 WMMA tiles --------------------
  {
    v16bf a = LOAD_A();
#pragma unroll
    for (int nt = 0; nt < 2; ++nt) {
      v16bf b = load_frag16(&sBF[6 + nt][lane][0]);
      union { v8f v; float f[8]; } acc;
#pragma unroll
      for (int j = 0; j < 8; ++j) acc.f[j] = 0.0f;
      acc.v = __builtin_amdgcn_wmma_f32_16x16x32_bf16(false, a, false, b,
                                                      (short)0, acc.v, false, false);
      const int col  = nt * 16 + cn;
      const float bias = sbm[col];
#pragma unroll
      for (int j = 0; j < 8; ++j)
        sX[wave][rb + j][col] = (bf16)fmaxf(acc.f[j] + bias, 0.0f);
    }
  }
  __syncthreads();

  // ---- LM head: [16x32] @ [32x80] + b -> 5 WMMA tiles (logits in LDS) ------
  {
    v16bf a = LOAD_A();
#pragma unroll
    for (int nt = 0; nt < 5; ++nt) {
      v16bf b = load_frag16(&sBF[8 + nt][lane][0]);
      union { v8f v; float f[8]; } acc;
#pragma unroll
      for (int j = 0; j < 8; ++j) acc.f[j] = 0.0f;
      acc.v = __builtin_amdgcn_wmma_f32_16x16x32_bf16(false, a, false, b,
                                                      (short)0, acc.v, false, false);
      const int col  = nt * 16 + cn;
      const float bias = sbl[col];
#pragma unroll
      for (int j = 0; j < 8; ++j) sF[wave][rb + j][col] = acc.f[j] + bias;
    }
  }
  __syncthreads();

  // ---- coalesced logits store: flat copy of 16x65 floats -------------------
  {
    const long gbase = tok_base * VOCAB;   // 16B-aligned when tok_base % 16 == 0
    if (full_tile) {
#pragma unroll
      for (int it = 0; it < 8; ++it) {
        const int i = (it * 32 + lane) * 4;          // 0..1020
        float4 w;
        w.x = sF[wave][(i    ) / VOCAB][(i    ) % VOCAB];
        w.y = sF[wave][(i + 1) / VOCAB][(i + 1) % VOCAB];
        w.z = sF[wave][(i + 2) / VOCAB][(i + 2) % VOCAB];
        w.w = sF[wave][(i + 3) / VOCAB][(i + 3) % VOCAB];
        *(float4*)(logits + gbase + i) = w;
      }
      if (lane < 16) {                               // tail: 1024..1039
        const int i = 1024 + lane;
        logits[gbase + i] = sF[wave][i / VOCAB][i % VOCAB];
      }
    } else {
      for (int i = lane; i < 16 * VOCAB; i += 32) {
        const int row = i / VOCAB, c = i - row * VOCAB;
        if (tok_base + row < n_tok) logits[gbase + i] = sF[wave][row][c];
      }
    }
  }

  // ---- NLL loss: vectorized row scan + deterministic reduction -------------
  float l = 0.0f;
  if (lane < 16) {
    const int row = lane;
    const long g = tok_base + row;
    if (g < n_tok) {
      const int t = tgt[g];
      float mx = -3.0e38f;
#pragma unroll
      for (int c4 = 0; c4 < 16; ++c4) {
        float4 v = *(const float4*)&sF[wave][row][c4 * 4];
        mx = fmaxf(mx, fmaxf(fmaxf(v.x, v.y), fmaxf(v.z, v.w)));
      }
      mx = fmaxf(mx, sF[wave][row][64]);
      float den = 0.0f;
#pragma unroll
      for (int c4 = 0; c4 < 16; ++c4) {
        float4 v = *(const float4*)&sF[wave][row][c4 * 4];
        den += __expf(v.x - mx) + __expf(v.y - mx) + __expf(v.z - mx) + __expf(v.w - mx);
      }
      den += __expf(sF[wave][row][64] - mx);
      l = -(sF[wave][row][t] - mx - __logf(den));
    }
  }
#pragma unroll
  for (int off = 16; off > 0; off >>= 1) l += __shfl_xor(l, off, 32);
  if (lane == 0) sWL[wave] = l;
  __syncthreads();
  if (tid == 0) {
    float tot = 0.0f;
#pragma unroll
    for (int w = 0; w < WAVES; ++w) tot += sWL[w];
    partials[blockIdx.x] = tot;
  }
#undef LOAD_A
}

__global__ __launch_bounds__(256) void loss_reduce(const float* __restrict__ partials, int n,
                                                   float* __restrict__ out, float inv_ntok)
{
  __shared__ float s[256];
  float acc = 0.0f;
  for (int i = threadIdx.x; i < n; i += 256) acc += partials[i];
  s[threadIdx.x] = acc;
  __syncthreads();
  for (int off = 128; off > 0; off >>= 1) {
    if ((int)threadIdx.x < off) s[threadIdx.x] += s[threadIdx.x + off];
    __syncthreads();
  }
  if (threadIdx.x == 0) *out = s[0] * inv_ntok;
}

extern "C" void kernel_launch(void* const* d_in, const int* in_sizes, int n_in,
                              void* d_out, int out_size, void* d_ws, size_t ws_size,
                              hipStream_t stream)
{
  const int*   idx     = (const int*)  d_in[0];
  const int*   targets = (const int*)  d_in[1];
  const float* tok_emb = (const float*)d_in[2];
  const float* pos_emb = (const float*)d_in[3];
  const float* Wq      = (const float*)d_in[4];
  const float* Wk      = (const float*)d_in[5];
  const float* Wv      = (const float*)d_in[6];
  const float* Wm      = (const float*)d_in[7];
  const float* bm      = (const float*)d_in[8];
  const float* Wl      = (const float*)d_in[9];
  const float* bl      = (const float*)d_in[10];

  const int n_tok  = in_sizes[0];             // B*T
  const int blocks = (n_tok + 63) / 64;       // 64 tokens per block (4 waves x 16)

  float* logits   = (float*)d_out;
  float* loss     = logits + (size_t)n_tok * VOCAB;
  float* partials = (float*)d_ws;             // blocks floats (<= 32 KB)

  fused_fwd<<<blocks, THREADS, 0, stream>>>(idx, targets, tok_emb, pos_emb,
                                            Wq, Wk, Wv, Wm, bm, Wl, bl,
                                            logits, partials, n_tok);
  loss_reduce<<<1, 256, 0, stream>>>(partials, blocks, loss, 1.0f / (float)n_tok);
}
